// DisentangledAttention_15049565405469
// MI455X (gfx1250) — matrix operand order repaired and checked by reading, not compiled, not run on hardware
//
#include <hip/hip_runtime.h>

typedef __bf16 bf16;
typedef __attribute__((ext_vector_type(16))) __bf16 v16bf;
typedef __attribute__((ext_vector_type(8)))  float  v8f;

struct U4 { unsigned int x, y, z, w; };   // 16-byte POD move unit

static constexpr int H_DIM  = 1024;
static constexpr int NHEAD  = 8;
static constexpr int HDIM   = 128;
static constexpr int FFDIM  = 3072;
static constexpr int BATCH  = 16;
static constexpr int SEQ    = 512;
static constexpr int MROWS  = BATCH * SEQ;   // 8192

// ---------------------------------------------------------------------------
// WMMA helpers (V_WMMA_F32_16X16X32_BF16), fragment loads per ISA 7.12.2
// ---------------------------------------------------------------------------
static __device__ __forceinline__ v8f wmma_bf16(v16bf a, v16bf b, v8f c) {
  return __builtin_amdgcn_wmma_f32_16x16x32_bf16(false, a, false, b, (short)0, c,
                                                 false, false);
}

// A fragment 16x32 (row-major [m][k] in LDS): lane m = lane&15, half = lane>>4
static __device__ __forceinline__ v16bf load_a_frag(const bf16* base, int ld, int koff) {
  int lane = threadIdx.x & 31;
  int m = lane & 15, half = lane >> 4;
  const bf16* r = base + m * ld + koff;
  union { v16bf v; U4 q[2]; } u;
  u.q[0] = *(const U4*)(r + half * 8);
  u.q[1] = *(const U4*)(r + 16 + half * 8);
  return u.v;
}

// B fragment 32x16 from transposed LDS tile [n][k]: lane n = lane&15,
// K = (lane>>4)*8 + 0..7 and 16 + (lane>>4)*8 + 0..7 (same as A)
static __device__ __forceinline__ v16bf load_b_frag(const bf16* baseT, int ld, int koff) {
  int lane = threadIdx.x & 31;
  int n = lane & 15, half = lane >> 4;
  const bf16* r = baseT + n * ld + koff;
  union { v16bf v; U4 q[2]; } u;
  u.q[0] = *(const U4*)(r + half * 8);
  u.q[1] = *(const U4*)(r + 16 + half * 8);
  return u.v;
}

// ---------------------------------------------------------------------------
// Async global->LDS copies (GLOBAL_LOAD_ASYNC_TO_LDS_B128, tracked by ASYNCcnt).
// Instruction offset is added to BOTH the LDS and global addresses (ISA 10.7).
// Flat generic pointer to a __shared__ object has the LDS offset in addr[31:0].
// ---------------------------------------------------------------------------
static __device__ __forceinline__ void async_copy_64B(const bf16* gsrc, bf16* ldst) {
  unsigned lo = (unsigned)(unsigned long long)ldst;
  unsigned long long ga = (unsigned long long)gsrc;
  asm volatile("global_load_async_to_lds_b128 %0, %1, off\n\t"
               "global_load_async_to_lds_b128 %0, %1, off offset:16\n\t"
               "global_load_async_to_lds_b128 %0, %1, off offset:32\n\t"
               "global_load_async_to_lds_b128 %0, %1, off offset:48"
               :: "v"(lo), "v"(ga) : "memory");
}
static __device__ __forceinline__ void async_copy_32B(const bf16* gsrc, bf16* ldst) {
  unsigned lo = (unsigned)(unsigned long long)ldst;
  unsigned long long ga = (unsigned long long)gsrc;
  asm volatile("global_load_async_to_lds_b128 %0, %1, off\n\t"
               "global_load_async_to_lds_b128 %0, %1, off offset:16"
               :: "v"(lo), "v"(ga) : "memory");
}
static __device__ __forceinline__ void wait_async() {
  asm volatile("s_wait_asynccnt 0x0" ::: "memory");
}

// ---------------------------------------------------------------------------
// f32 -> bf16 convert (plain) and tiled transpose-convert W[K][N] -> Wt[N][K]
// ---------------------------------------------------------------------------
__global__ __launch_bounds__(256) void cvt_kernel(const float* __restrict__ in,
                                                  bf16* __restrict__ out, int n) {
  int i = blockIdx.x * 256 + threadIdx.x;
  if (i < n) out[i] = (bf16)in[i];
}

__global__ __launch_bounds__(256) void cvt_t_kernel(const float* __restrict__ in,
                                                    bf16* __restrict__ out,
                                                    int K, int N) {
  __shared__ float tile[32][33];
  int nb = blockIdx.x * 32, kb = blockIdx.y * 32;
  int tx = threadIdx.x & 31, ty = threadIdx.x >> 5;   // ty in 0..7
#pragma unroll
  for (int i = 0; i < 4; ++i) {
    int k = kb + ty + i * 8;
    tile[ty + i * 8][tx] = in[(size_t)k * N + nb + tx];
  }
  __syncthreads();
#pragma unroll
  for (int i = 0; i < 4; ++i) {
    int n = nb + ty + i * 8;
    out[(size_t)n * K + kb + tx] = (bf16)tile[tx][ty + i * 8];
  }
}

// ---------------------------------------------------------------------------
// LayerNorm over H=1024; optional "+pos_emb[row%SEQ]"; optional f32/bf16 out
// ---------------------------------------------------------------------------
__global__ __launch_bounds__(256) void ln_kernel(const float* __restrict__ src,
                                                 const float* __restrict__ pos,
                                                 const float* __restrict__ gam,
                                                 const float* __restrict__ bet,
                                                 float* __restrict__ outF,
                                                 bf16* __restrict__ outB) {
  int row = blockIdx.x;
  int t = threadIdx.x;
  const float* sr = src + (size_t)row * H_DIM;
  const float* pr = pos ? pos + (size_t)(row & (SEQ - 1)) * H_DIM : nullptr;
  float vals[4];
  float lsum = 0.f, lsq = 0.f;
#pragma unroll
  for (int i = 0; i < 4; ++i) {
    int c = t + i * 256;
    float v = sr[c] + (pos ? pr[c] : 0.f);
    vals[i] = v;
    lsum += v;
    lsq += v * v;
  }
#pragma unroll
  for (int o = 16; o >= 1; o >>= 1) {
    lsum += __shfl_xor(lsum, o, 32);
    lsq  += __shfl_xor(lsq,  o, 32);
  }
  __shared__ float s1[8], s2[8];
  int wave = t >> 5, lane = t & 31;
  if (lane == 0) { s1[wave] = lsum; s2[wave] = lsq; }
  __syncthreads();
  float sum = 0.f, sq = 0.f;
#pragma unroll
  for (int w = 0; w < 8; ++w) { sum += s1[w]; sq += s2[w]; }
  float mean = sum * (1.f / H_DIM);
  float var = sq * (1.f / H_DIM) - mean * mean;
  float inv = rsqrtf(var + 1e-7f);
#pragma unroll
  for (int i = 0; i < 4; ++i) {
    int c = t + i * 256;
    float o = (vals[i] - mean) * inv * gam[c] + bet[c];
    if (outF) outF[(size_t)row * H_DIM + c] = o;
    if (outB) outB[(size_t)row * H_DIM + c] = (bf16)o;
  }
}

// ---------------------------------------------------------------------------
// Tiled WMMA GEMM:  C[M,N] = A[M,K](bf16) x Wt[N,K](bf16, pre-transposed)
//   EPI 0: +bias(optional), scatter bf16 to [B,NH,S,HD]    (Q,K projections)
//   EPI 1: +bias +residual -> f32 row-major                (pre-LayerNorm sum)
//   EPI 2: +bias, exact GELU -> bf16 row-major             (FF intermediate)
//   EPI 3: +bias, scatter bf16 to [B,NH,HD,S] (transposed) (V projection)
// Block tile 128x128, BK=64, 256 threads (8 waves, 4x2), double-buffered LDS,
// async global->LDS staging overlapped with WMMA compute.
// ---------------------------------------------------------------------------
static __device__ __forceinline__ void stage_rows(const bf16* __restrict__ src, int ld,
                                                  int rowbase, int kbase, bf16* dst,
                                                  int t) {
  int row = t >> 1;
  int kc = (t & 1) * 32;
  async_copy_64B(src + (size_t)(rowbase + row) * ld + kbase + kc, dst + row * 64 + kc);
}

template <int EPI>
__global__ __launch_bounds__(256) void gemm_kernel(const bf16* __restrict__ A,
                                                   const bf16* __restrict__ Wt,
                                                   const float* __restrict__ bias,
                                                   const float* __restrict__ res,
                                                   float* __restrict__ outF,
                                                   bf16* __restrict__ outB,
                                                   int N, int K) {
  __shared__ __align__(16) bf16 As[2][128 * 64];
  __shared__ __align__(16) bf16 Bs[2][128 * 64];   // [n][k]

  int mbase = blockIdx.x * 128;
  int nbase = blockIdx.y * 128;
  int t = threadIdx.x;
  int wave = t >> 5;
  int wm = wave & 3, wn = wave >> 2;

  v8f acc[2][4];
#pragma unroll
  for (int i = 0; i < 2; ++i)
#pragma unroll
    for (int j = 0; j < 4; ++j) acc[i][j] = (v8f){0.f, 0.f, 0.f, 0.f, 0.f, 0.f, 0.f, 0.f};

  int nkt = K / 64;
  stage_rows(A,  K, mbase, 0, As[0], t);
  stage_rows(Wt, K, nbase, 0, Bs[0], t);
  wait_async();
  __syncthreads();

  for (int kt = 0; kt < nkt; ++kt) {
    int cur = kt & 1, nxt = cur ^ 1;
    if (kt + 1 < nkt) {
      stage_rows(A,  K, mbase, (kt + 1) * 64, As[nxt], t);
      stage_rows(Wt, K, nbase, (kt + 1) * 64, Bs[nxt], t);
    }
#pragma unroll
    for (int kc = 0; kc < 64; kc += 32) {
      v16bf af0 = load_a_frag(&As[cur][(wm * 32) * 64], 64, kc);
      v16bf af1 = load_a_frag(&As[cur][(wm * 32 + 16) * 64], 64, kc);
#pragma unroll
      for (int j = 0; j < 4; ++j) {
        v16bf bfj = load_b_frag(&Bs[cur][(wn * 64 + j * 16) * 64], 64, kc);
        acc[0][j] = wmma_bf16(af0, bfj, acc[0][j]);
        acc[1][j] = wmma_bf16(af1, bfj, acc[1][j]);
      }
    }
    wait_async();
    __syncthreads();
  }

  int lane = t & 31;
  int cl = lane & 15, rh = (lane >> 4) * 8;
#pragma unroll
  for (int i = 0; i < 2; ++i) {
#pragma unroll
    for (int j = 0; j < 4; ++j) {
      int col = nbase + wn * 64 + j * 16 + cl;
      int row0 = mbase + wm * 32 + i * 16 + rh;
      float bv = bias ? bias[col] : 0.f;
#pragma unroll
      for (int v = 0; v < 8; ++v) {
        int row = row0 + v;
        float val = acc[i][j][v] + bv;
        if (EPI == 0) {
          int b = row >> 9, s = row & (SEQ - 1);
          int head = col >> 7, d = col & (HDIM - 1);
          outB[((size_t)(b * NHEAD + head) * SEQ + s) * HDIM + d] = (bf16)val;
        } else if (EPI == 1) {
          val += res[(size_t)row * N + col];
          outF[(size_t)row * N + col] = val;
        } else if (EPI == 2) {
          float g = 0.5f * val * (1.f + erff(val * 0.70710678118654752f));
          outB[(size_t)row * N + col] = (bf16)g;
        } else {  // EPI 3: V scattered transposed [B,NH,HD,S]
          int b = row >> 9, s = row & (SEQ - 1);
          int head = col >> 7, d = col & (HDIM - 1);
          outB[((size_t)(b * NHEAD + head) * HDIM + d) * SEQ + s] = (bf16)val;
        }
      }
    }
  }
}

// ---------------------------------------------------------------------------
// Fused attention per (b, head, 32-query tile):
//   scores = (Q/sqrt(HD)) K^T -> bf16 LDS -> softmax in place -> P V -> ctx
// q,k in [B,NH,S,HD]; v pre-transposed [B,NH,HD,S] so V^T stages contiguously.
// 256 threads (8 waves). LDS = 8KB Q + 16KB K/V^T + 32KB scores = 56KB.
// ---------------------------------------------------------------------------
__global__ __launch_bounds__(256) void attn_kernel(const bf16* __restrict__ q,
                                                   const bf16* __restrict__ k,
                                                   const bf16* __restrict__ v,
                                                   bf16* __restrict__ ctx) {
  constexpr int QT = 32, KT = 64;
  __shared__ __align__(16) bf16 Qs[QT * HDIM];    // 8 KB
  __shared__ __align__(16) bf16 Ks[KT * HDIM];    // 16 KB (reused as V^T [d][key])
  __shared__ __align__(16) bf16 Sb[QT * SEQ];     // 32 KB scores -> probs

  int qt = blockIdx.x;
  int head = blockIdx.y;
  int b = blockIdx.z;
  int t = threadIdx.x, wave = t >> 5, lane = t & 31;
  size_t headoff = (size_t)(b * NHEAD + head) * SEQ * HDIM;
  int qbase = qt * QT;

  {  // async-load Q tile (waited inside phase-A's first iteration)
    int row = t >> 3, dc = (t & 7) * 16;
    async_copy_32B(q + headoff + (size_t)(qbase + row) * HDIM + dc, &Qs[row * HDIM + dc]);
  }

  const float scale = 0.08838834764831845f;  // 1/sqrt(128)
  int rt = wave & 1, ct = wave >> 1;         // wave's 16x16 tile within 32x64 block

  // ---- phase A: scores ----
  for (int kt = 0; kt < SEQ / KT; ++kt) {
    __syncthreads();
    {  // stage K tile 64x128 (row-major [key][d] == B^T layout [n][k])
      int row = t >> 2, dc = (t & 3) * 32;
      async_copy_64B(k + headoff + (size_t)(kt * KT + row) * HDIM + dc,
                     &Ks[row * HDIM + dc]);
    }
    wait_async();
    __syncthreads();
    // two independent accumulator chains over HD=128
    v8f a0 = (v8f){0.f, 0.f, 0.f, 0.f, 0.f, 0.f, 0.f, 0.f};
    v8f a1 = a0;
#pragma unroll
    for (int kc = 0; kc < HDIM; kc += 64) {
      v16bf af0 = load_a_frag(&Qs[(rt * 16) * HDIM], HDIM, kc);
      v16bf bf0 = load_b_frag(&Ks[(ct * 16) * HDIM], HDIM, kc);
      v16bf af1 = load_a_frag(&Qs[(rt * 16) * HDIM], HDIM, kc + 32);
      v16bf bf1 = load_b_frag(&Ks[(ct * 16) * HDIM], HDIM, kc + 32);
      a0 = wmma_bf16(af0, bf0, a0);
      a1 = wmma_bf16(af1, bf1, a1);
    }
    int ccol = kt * KT + ct * 16 + (lane & 15);
    int r0 = rt * 16 + (lane >> 4) * 8;
#pragma unroll
    for (int vv = 0; vv < 8; ++vv)
      Sb[(r0 + vv) * SEQ + ccol] = (bf16)((a0[vv] + a1[vv]) * scale);
  }
  __syncthreads();

  // ---- phase B: softmax, 8 lanes per row, in place ----
  {
    int row = t >> 3, g = t & 7;
    bf16* sr = &Sb[row * SEQ];
    float mx = -1e30f;
    for (int c = g * 64; c < g * 64 + 64; ++c) mx = fmaxf(mx, (float)sr[c]);
#pragma unroll
    for (int o = 4; o >= 1; o >>= 1) mx = fmaxf(mx, __shfl_xor(mx, o, 8));
    float sum = 0.f;
    for (int c = g * 64; c < g * 64 + 64; ++c) sum += __expf((float)sr[c] - mx);
#pragma unroll
    for (int o = 4; o >= 1; o >>= 1) sum += __shfl_xor(sum, o, 8);
    float inv = 1.f / sum;
    for (int c = g * 64; c < g * 64 + 64; ++c)
      sr[c] = (bf16)(__expf((float)sr[c] - mx) * inv);
  }

  // ---- phase C: ctx = P V ----
  v8f o0 = (v8f){0.f, 0.f, 0.f, 0.f, 0.f, 0.f, 0.f, 0.f};
  v8f o1 = o0;
  for (int kt = 0; kt < SEQ / KT; ++kt) {
    __syncthreads();
    {  // stage V^T tile [d][key] from pre-transposed v: contiguous copies
      int row = t >> 1, kc2 = (t & 1) * 32;   // row = d in 0..127
      async_copy_64B(v + headoff + (size_t)row * SEQ + kt * KT + kc2,
                     &Ks[row * KT + kc2]);
    }
    wait_async();
    __syncthreads();
#pragma unroll
    for (int kc = 0; kc < KT; kc += 32) {
      v16bf af = load_a_frag(&Sb[(rt * 16) * SEQ], SEQ, kt * KT + kc);
      v16bf b0 = load_b_frag(&Ks[(ct * 16) * KT], KT, kc);
      v16bf b1 = load_b_frag(&Ks[((ct + 4) * 16) * KT], KT, kc);
      o0 = wmma_bf16(af, b0, o0);
      o1 = wmma_bf16(af, b1, o1);
    }
  }
  {  // write ctx as bf16 [B, S, H]
    int r0 = rt * 16 + (lane >> 4) * 8;
    int d0 = ct * 16 + (lane & 15);
    int d1 = (ct + 4) * 16 + (lane & 15);
#pragma unroll
    for (int vv = 0; vv < 8; ++vv) {
      int srow = qbase + r0 + vv;
      size_t base = ((size_t)(b * SEQ + srow)) * H_DIM + head * HDIM;
      ctx[base + d0] = (bf16)o0[vv];
      ctx[base + d1] = (bf16)o1[vv];
    }
  }
}

// ---------------------------------------------------------------------------
// Host side
// ---------------------------------------------------------------------------
extern "C" void kernel_launch(void* const* d_in, const int* in_sizes, int n_in,
                              void* d_out, int out_size, void* d_ws, size_t ws_size,
                              hipStream_t stream) {
  const float* x      = (const float*)d_in[0];
  const float* pos    = (const float*)d_in[1];
  const float* emb_g  = (const float*)d_in[2];
  const float* emb_b  = (const float*)d_in[3];
  const float* Wq     = (const float*)d_in[4];
  const float* Wk     = (const float*)d_in[5];
  const float* Wv     = (const float*)d_in[6];
  const float* q_bias = (const float*)d_in[7];
  const float* v_bias = (const float*)d_in[8];
  const float* Wo     = (const float*)d_in[9];
  const float* bo     = (const float*)d_in[10];
  const float* ln1_g  = (const float*)d_in[11];
  const float* ln1_b  = (const float*)d_in[12];
  const float* Wi     = (const float*)d_in[13];
  const float* bi     = (const float*)d_in[14];
  const float* Wf     = (const float*)d_in[15];
  const float* bfv    = (const float*)d_in[16];
  const float* ln2_g  = (const float*)d_in[17];
  const float* ln2_b  = (const float*)d_in[18];

  char* ws = (char*)d_ws;
  size_t off = 0;
  auto take = [&](size_t bytes) -> char* {
    char* p = ws + off;
    off += (bytes + 255) & ~(size_t)255;
    return p;
  };
  float* hF   = (float*)take((size_t)MROWS * H_DIM * 4);
  bf16* hB    = (bf16*)take((size_t)MROWS * H_DIM * 2);
  bf16* WqT   = (bf16*)take((size_t)H_DIM * H_DIM * 2);
  bf16* WkT   = (bf16*)take((size_t)H_DIM * H_DIM * 2);
  bf16* WvT   = (bf16*)take((size_t)H_DIM * H_DIM * 2);
  bf16* WoT   = (bf16*)take((size_t)H_DIM * H_DIM * 2);
  bf16* WiT   = (bf16*)take((size_t)H_DIM * FFDIM * 2);
  bf16* WfT   = (bf16*)take((size_t)FFDIM * H_DIM * 2);
  bf16* qB    = (bf16*)take((size_t)MROWS * H_DIM * 2);
  bf16* kB    = (bf16*)take((size_t)MROWS * H_DIM * 2);
  bf16* vB    = (bf16*)take((size_t)MROWS * H_DIM * 2);   // [B,NH,HD,S]
  bf16* ctxB  = (bf16*)take((size_t)MROWS * H_DIM * 2);
  float* pre  = (float*)take((size_t)MROWS * H_DIM * 4);
  float* attnF = (float*)take((size_t)MROWS * H_DIM * 4);
  bf16* attnB  = (bf16*)take((size_t)MROWS * H_DIM * 2);
  bf16* ffB    = (bf16*)take((size_t)MROWS * FFDIM * 2);

  // 1) weights -> bf16, transposed to [N][K]
  dim3 tHH(H_DIM / 32, H_DIM / 32);
  cvt_t_kernel<<<tHH, 256, 0, stream>>>(Wq, WqT, H_DIM, H_DIM);
  cvt_t_kernel<<<tHH, 256, 0, stream>>>(Wk, WkT, H_DIM, H_DIM);
  cvt_t_kernel<<<tHH, 256, 0, stream>>>(Wv, WvT, H_DIM, H_DIM);
  cvt_t_kernel<<<tHH, 256, 0, stream>>>(Wo, WoT, H_DIM, H_DIM);
  cvt_t_kernel<<<dim3(FFDIM / 32, H_DIM / 32), 256, 0, stream>>>(Wi, WiT, H_DIM, FFDIM);
  cvt_t_kernel<<<dim3(H_DIM / 32, FFDIM / 32), 256, 0, stream>>>(Wf, WfT, FFDIM, H_DIM);

  // 2) h = LN(x + pos_emb)
  ln_kernel<<<MROWS, 256, 0, stream>>>(x, pos, emb_g, emb_b, hF, hB);

  // 3) Q/K/V projections (Q,K -> [B,NH,S,HD]; V -> [B,NH,HD,S])
  dim3 gHH(MROWS / 128, H_DIM / 128);
  gemm_kernel<0><<<gHH, 256, 0, stream>>>(hB, WqT, q_bias, nullptr, nullptr, qB, H_DIM, H_DIM);
  gemm_kernel<0><<<gHH, 256, 0, stream>>>(hB, WkT, nullptr, nullptr, nullptr, kB, H_DIM, H_DIM);
  gemm_kernel<3><<<gHH, 256, 0, stream>>>(hB, WvT, v_bias, nullptr, nullptr, vB, H_DIM, H_DIM);

  // 4) attention
  attn_kernel<<<dim3(SEQ / 32, NHEAD, BATCH), 256, 0, stream>>>(qB, kB, vB, ctxB);

  // 5) pre = ctx @ Wo + bo + h ; attn_out = LN(pre)
  gemm_kernel<1><<<gHH, 256, 0, stream>>>(ctxB, WoT, bo, hF, pre, nullptr, H_DIM, H_DIM);
  ln_kernel<<<MROWS, 256, 0, stream>>>(pre, nullptr, ln1_g, ln1_b, attnF, attnB);

  // 6) ff = gelu(attn_out @ Wi + bi)
  dim3 gHFd(MROWS / 128, FFDIM / 128);
  gemm_kernel<2><<<gHFd, 256, 0, stream>>>(attnB, WiT, bi, nullptr, nullptr, ffB, FFDIM, H_DIM);

  // 7) pre = ff @ Wf + bf + attn_out ; out = LN(pre)
  gemm_kernel<1><<<gHH, 256, 0, stream>>>(ffB, WfT, bfv, attnF, pre, nullptr, H_DIM, FFDIM);
  ln_kernel<<<MROWS, 256, 0, stream>>>(pre, nullptr, ln2_g, ln2_b, (float*)d_out, nullptr);
}
